// myBadTransfomerBlock_8280696946837
// MI455X (gfx1250) — matrix operand order repaired and checked by reading, not compiled, not run on hardware
//
#include <hip/hip_runtime.h>
#include <hip/hip_bf16.h>

typedef __bf16 bf16_t;
typedef __attribute__((ext_vector_type(16))) __bf16 v16bf;
typedef __attribute__((ext_vector_type(8)))  float  v8f;
typedef __attribute__((ext_vector_type(4)))  __bf16 v4bf;

#define TDIM 4096
#define TBATCH 4
#define TSEQ 2048
#define TMX (TBATCH*TSEQ)   /* 8192 rows of x flattened */

// ---------------------------------------------------------------------------
// CDNA5 async global->LDS copy (ASYNCcnt-tracked DMA, no VGPR round-trip).
// INST_OFFSET is added to BOTH the LDS and the global address (ISA 08 §4.4).
// LDS operand = byte offset within workgroup LDS = low 32 bits of generic ptr.
// ---------------------------------------------------------------------------
template<int OFF>
__device__ __forceinline__ void async_ld_b128(const void* lds_ptr, const void* gptr) {
  asm volatile("global_load_async_to_lds_b128 %0, %1, off offset:%2"
               :: "v"((unsigned)(unsigned long long)lds_ptr),
                  "v"((unsigned long long)gptr),
                  "i"(OFF)
               : "memory");
}
__device__ __forceinline__ void wait_asynccnt0() {
  asm volatile("s_wait_asynccnt 0x0" ::: "memory");
}

// ---------------------------------------------------------------------------
// fp32 -> bf16 cast (vectorized 4-wide)
// ---------------------------------------------------------------------------
__global__ __launch_bounds__(256)
void cast_f32_to_bf16(const float* __restrict__ in, bf16_t* __restrict__ out, int n4) {
  int i = blockIdx.x * blockDim.x + threadIdx.x;
  if (i >= n4) return;
  float4 v = ((const float4*)in)[i];
  v4bf o;
  o.x = (bf16_t)v.x; o.y = (bf16_t)v.y; o.z = (bf16_t)v.z; o.w = (bf16_t)v.w;
  ((v4bf*)out)[i] = o;
}

// ---------------------------------------------------------------------------
// bf16 WMMA GEMM:  C[m,n] = sum_k A[m,k] * Bt[n,k]
// Block tile 128x256, K-step 32. 8 waves: 2(M) x 4(N); wave tile 64x64 =
// 4x4 accumulator tiles (128 acc VGPRs), 16 v_wmma per K-step vs 16 ds_load.
// Global->LDS staging via global_load_async_to_lds_b128 (double-buffered,
// one s_wait_asynccnt + barrier per K-step). LDS rows padded to 40 halves.
// ---------------------------------------------------------------------------
#define BM 128
#define BN 256
#define BK 32
#define LDSS 40

union FragAB { v16bf v; uint4 q[2]; };

enum { EPI_F32 = 0, EPI_BF16 = 1, EPI_SILU_ADDX = 2 };

template<int EPI>
__global__ __launch_bounds__(256)
void wmma_gemm(const bf16_t* __restrict__ A,  long long sAb, int lda,
               const bf16_t* __restrict__ Bt, long long sBb, int ldb,
               void* __restrict__ Cv,         long long sCb, int ldc,
               const float* __restrict__ Xadd, int K)
{
  // per buffer: A block 128 rows + B block 256 rows, stride 40 halves
  __shared__ __align__(16) bf16_t smem[2 * (BM + BN) * LDSS];   // 60 KB

  const int t     = threadIdx.x;
  const int lane  = t & 31;
  const int wid   = t >> 5;
  const int waveM = wid >> 2;        // 0..1
  const int waveN = wid & 3;         // 0..3
  const int l16   = lane & 15;
  const int lhi   = lane >> 4;       // 0/1

  const int bm0 = blockIdx.y * BM;
  const int bn0 = blockIdx.x * BN;
  A  += (long long)blockIdx.z * sAb;
  Bt += (long long)blockIdx.z * sBb;

  const int BUF  = (BM + BN) * LDSS; // elements per double-buffer slot
  const int BOFF = BM * LDSS;        // B block offset inside a slot

  // --- staging assignment ---
  // A: thread -> (row ra, 16-half chunk ha): 2 x b128 async
  const int ra = t >> 1;             // 0..127
  const int ha = t & 1;              // 0/1
  const bf16_t* gA = A  + (size_t)(bm0 + ra) * lda + ha * 16;
  // B: thread -> row t (0..255): 4 x b128 async
  const bf16_t* gB = Bt + (size_t)(bn0 + t) * ldb;

  // prologue: stage k-block 0 into buffer 0
  {
    const bf16_t* lA = &smem[ra * LDSS + ha * 16];
    const bf16_t* lB = &smem[BOFF + t * LDSS];
    async_ld_b128<0>(lA, gA);
    async_ld_b128<16>(lA, gA);
    async_ld_b128<0>(lB, gB);
    async_ld_b128<16>(lB, gB);
    async_ld_b128<32>(lB, gB);
    async_ld_b128<48>(lB, gB);
  }
  wait_asynccnt0();
  __syncthreads();

  v8f acc[4][4];
#pragma unroll
  for (int i = 0; i < 4; ++i)
#pragma unroll
    for (int j = 0; j < 4; ++j)
#pragma unroll
      for (int e = 0; e < 8; ++e) acc[i][j][e] = 0.0f;

  const int nk = K / BK;
  int buf = 0;
  for (int kb = 0; kb < nk; ++kb) {
    // kick off async DMA of the next k-slab into the other buffer
    if (kb + 1 < nk) {
      const int nb = (buf ^ 1) * BUF;
      const bf16_t* pa = gA + (size_t)(kb + 1) * BK;
      const bf16_t* pb = gB + (size_t)(kb + 1) * BK;
      const bf16_t* lA = &smem[nb + ra * LDSS + ha * 16];
      const bf16_t* lB = &smem[nb + BOFF + t * LDSS];
      async_ld_b128<0>(lA, pa);
      async_ld_b128<16>(lA, pa);
      async_ld_b128<0>(lB, pb);
      async_ld_b128<16>(lB, pb);
      async_ld_b128<32>(lB, pb);
      async_ld_b128<48>(lB, pb);
    }

    const bf16_t* base = smem + buf * BUF;
    FragAB b[4];
#pragma unroll
    for (int j = 0; j < 4; ++j) {             // B frag: lane = column n = l16
      const int rn = waveN * 64 + j * 16 + l16;
      const bf16_t* p = base + BOFF + rn * LDSS + lhi * 16;  // K 0..15 / 16..31
      b[j].q[0] = *(const uint4*)(p);
      b[j].q[1] = *(const uint4*)(p + 8);
    }
#pragma unroll
    for (int i = 0; i < 4; ++i) {             // A frag: lane = row m = l16
      const int rm = waveM * 64 + i * 16 + l16;
      const bf16_t* p = base + rm * LDSS + lhi * 16;         // K 0..15 / 16..31
      FragAB a;
      a.q[0] = *(const uint4*)(p);
      a.q[1] = *(const uint4*)(p + 8);
#pragma unroll
      for (int j = 0; j < 4; ++j)
        acc[i][j] = __builtin_amdgcn_wmma_f32_16x16x32_bf16(
            false, a.v, false, b[j].v, (short)0, acc[i][j], false, false);
    }

    if (kb + 1 < nk) wait_asynccnt0();  // next buffer's DMA landed (this wave)
    __syncthreads();                    // all waves done reading + DMAs landed
    buf ^= 1;
  }

  // ---- epilogue: C fragment layout (VGPR r: M=r / r+8; N = lane%16) ----
  const long long cb = (long long)blockIdx.z * sCb;
#pragma unroll
  for (int i = 0; i < 4; ++i) {
#pragma unroll
    for (int j = 0; j < 4; ++j) {
#pragma unroll
      for (int r = 0; r < 8; ++r) {
        const int m = bm0 + waveM * 64 + i * 16 + r + (lhi << 3);
        const int n = bn0 + waveN * 64 + j * 16 + l16;
        const size_t idx = (size_t)m * ldc + n;
        const float val = acc[i][j][r];
        if (EPI == EPI_F32) {
          ((float*)Cv)[cb + idx] = val;
        } else if (EPI == EPI_BF16) {
          ((bf16_t*)Cv)[cb + idx] = (bf16_t)val;
        } else { // EPI_SILU_ADDX: z = bf16(x + silu(val))
          const float sil = val / (1.0f + __expf(-val));
          ((bf16_t*)Cv)[cb + idx] = (bf16_t)(Xadd[idx] + sil);
        }
      }
    }
  }
}

// ---------------------------------------------------------------------------
// Column softmax over sqT[b][e][d]: softmax over e for fixed (b,d).
// Output Pt[b][e][d] = softmax(sq[b][d][:])[e] in bf16 (= B^T operand).
// ---------------------------------------------------------------------------
__global__ __launch_bounds__(1024)
void col_softmax(const float* __restrict__ sq, bf16_t* __restrict__ pt) {
  const int b = blockIdx.y;
  const int c = threadIdx.x;   // 0..63
  const int r = threadIdx.y;   // 0..15
  const int d = blockIdx.x * 64 + c;
  const float* base = sq + (size_t)b * TDIM * TDIM;
  bf16_t*      out  = pt + (size_t)b * TDIM * TDIM;

  float m = -3.402823466e38f, s = 0.0f;
  for (int e = r; e < TDIM; e += 16) {
    const float v = base[(size_t)e * TDIM + d];
    const float nm = fmaxf(m, v);
    s = s * __expf(m - nm) + __expf(v - nm);
    m = nm;
  }
  __shared__ float sm[16][64], ss[16][64];
  sm[r][c] = m; ss[r][c] = s;
  __syncthreads();
  if (r == 0) {
    float M = sm[0][c], S = ss[0][c];
    for (int k = 1; k < 16; ++k) {
      const float nm = fmaxf(M, sm[k][c]);
      S = S * __expf(M - nm) + ss[k][c] * __expf(sm[k][c] - nm);
      M = nm;
    }
    sm[0][c] = M; ss[0][c] = 1.0f / S;
  }
  __syncthreads();
  const float M = sm[0][c], rs = ss[0][c];
  for (int e = r; e < TDIM; e += 16) {
    const float v = base[(size_t)e * TDIM + d];
    out[(size_t)e * TDIM + d] = (bf16_t)(__expf(v - M) * rs);
  }
}

// ---------------------------------------------------------------------------
// Row softmax: one block per row of xO[8192][4096]; writes bf16.
// ---------------------------------------------------------------------------
__global__ __launch_bounds__(256)
void row_softmax(const float* __restrict__ in, bf16_t* __restrict__ out) {
  __shared__ float red[256];
  const size_t row = blockIdx.x;
  const float* p = in + row * TDIM;
  const int t = threadIdx.x;

  float m = -3.402823466e38f;
  for (int i = t; i < TDIM; i += 256) m = fmaxf(m, p[i]);
  red[t] = m; __syncthreads();
  for (int s = 128; s > 0; s >>= 1) { if (t < s) red[t] = fmaxf(red[t], red[t + s]); __syncthreads(); }
  m = red[0];
  __syncthreads();

  float sum = 0.0f;
  for (int i = t; i < TDIM; i += 256) sum += __expf(p[i] - m);
  red[t] = sum; __syncthreads();
  for (int s = 128; s > 0; s >>= 1) { if (t < s) red[t] += red[t + s]; __syncthreads(); }
  const float rs = 1.0f / red[0];

  bf16_t* o = out + row * TDIM;
  for (int i = t; i < TDIM; i += 256) o[i] = (bf16_t)(__expf(p[i] - m) * rs);
}

// ---------------------------------------------------------------------------
extern "C" void kernel_launch(void* const* d_in, const int* in_sizes, int n_in,
                              void* d_out, int out_size, void* d_ws, size_t ws_size,
                              hipStream_t stream)
{
  (void)in_sizes; (void)n_in; (void)out_size; (void)ws_size;

  const float* x  = (const float*)d_in[0];
  const float* W1 = (const float*)d_in[1];
  const float* W2 = (const float*)d_in[2];
  const float* W3 = (const float*)d_in[3];
  const float* W4 = (const float*)d_in[4];
  const float* Wo = (const float*)d_in[5];

  char* ws = (char*)d_ws;
  const size_t NX = (size_t)TMX * TDIM;     // 33.55M
  const size_t NW = (size_t)TDIM * TDIM;    // 16.78M

  // workspace layout (with overlays of dead buffers); total 672 MB
  bf16_t* xh  = (bf16_t*)(ws);                         // 64 MB
  bf16_t* W1h = (bf16_t*)(ws + NX * 2);                // 32 MB each
  bf16_t* W2h = W1h + NW;
  bf16_t* W3h = W2h + NW;
  bf16_t* W4h = W3h + NW;
  bf16_t* Woh = W4h + NW;
  bf16_t* xAT = Woh + NW;                              // [DIM][MX] bf16, 64 MB
  bf16_t* xBT = xAT + NX;                              // 64 MB
  bf16_t* xC  = xBT + NX;                              // [MX][DIM] bf16, 64 MB
  float*  sqT = (float*)((char*)(xC + NX));            // [B][DIM][DIM] f32, 256 MB
  bf16_t* Pt  = xAT;                                   // overlay (xAT/xBT dead)
  float*  xO  = sqT;                                   // overlay (sqT dead)
  bf16_t* xOs = (bf16_t*)((char*)sqT + NX * sizeof(float)); // after xO's 128 MB
  bf16_t* Z   = (bf16_t*)(ws);                         // overlay xh (dead)

  const dim3 blk(256);

  // 1) casts to bf16
  {
    const int n4x = (int)(NX / 4), n4w = (int)(NW / 4);
    cast_f32_to_bf16<<<(n4x + 255) / 256, blk, 0, stream>>>(x,  xh,  n4x);
    cast_f32_to_bf16<<<(n4w + 255) / 256, blk, 0, stream>>>(W1, W1h, n4w);
    cast_f32_to_bf16<<<(n4w + 255) / 256, blk, 0, stream>>>(W2, W2h, n4w);
    cast_f32_to_bf16<<<(n4w + 255) / 256, blk, 0, stream>>>(W3, W3h, n4w);
    cast_f32_to_bf16<<<(n4w + 255) / 256, blk, 0, stream>>>(W4, W4h, n4w);
    cast_f32_to_bf16<<<(n4w + 255) / 256, blk, 0, stream>>>(Wo, Woh, n4w);
  }

  // 2) xA^T = W1 @ x^T  (stored row-major [e][b*S+s], i.e. already transposed)
  wmma_gemm<EPI_BF16><<<dim3(TMX / BN, TDIM / BM, 1), blk, 0, stream>>>(
      W1h, 0, TDIM, xh, 0, TDIM, xAT, 0, TMX, nullptr, TDIM);
  // 3) xB^T = W2 @ x^T
  wmma_gemm<EPI_BF16><<<dim3(TMX / BN, TDIM / BM, 1), blk, 0, stream>>>(
      W2h, 0, TDIM, xh, 0, TDIM, xBT, 0, TMX, nullptr, TDIM);
  // 4) xC = x @ W3^T
  wmma_gemm<EPI_BF16><<<dim3(TDIM / BN, TMX / BM, 1), blk, 0, stream>>>(
      xh, 0, TDIM, W3h, 0, TDIM, xC, 0, TDIM, nullptr, TDIM);

  // 5) sq^T[b][e][d] = sum_s xB[b,s,e]*xA[b,s,d]   (batched; K = SEQ window)
  wmma_gemm<EPI_F32><<<dim3(TDIM / BN, TDIM / BM, TBATCH), blk, 0, stream>>>(
      xBT, TSEQ, TMX, xAT, TSEQ, TMX, sqT, (long long)TDIM * TDIM, TDIM,
      nullptr, TSEQ);

  // 6) softmax over original last axis == column softmax of sq^T -> Pt bf16
  col_softmax<<<dim3(TDIM / 64, TBATCH), dim3(64, 16), 0, stream>>>(sqT, Pt);

  // 7) xO[b] = xC[b] @ P[b]   (Bt = Pt, exactly the transposed-P layout)
  wmma_gemm<EPI_F32><<<dim3(TDIM / BN, TSEQ / BM, TBATCH), blk, 0, stream>>>(
      xC, (long long)TSEQ * TDIM, TDIM, Pt, (long long)TDIM * TDIM, TDIM,
      xO, (long long)TSEQ * TDIM, TDIM, nullptr, TDIM);

  // 8) row softmax -> bf16
  row_softmax<<<TMX, blk, 0, stream>>>(xO, xOs);

  // 9) Z = bf16( x + silu(xOs @ W4^T) )   (fused epilogue)
  wmma_gemm<EPI_SILU_ADDX><<<dim3(TDIM / BN, TMX / BM, 1), blk, 0, stream>>>(
      xOs, 0, TDIM, W4h, 0, TDIM, Z, 0, TDIM, x, TDIM);

  // 10) out = Z @ Wo^T  (fp32 straight to d_out)
  wmma_gemm<EPI_F32><<<dim3(TDIM / BN, TMX / BM, 1), blk, 0, stream>>>(
      Z, 0, TDIM, Woh, 0, TDIM, d_out, 0, TDIM, nullptr, TDIM);
}
